// MyLSTM_26113401159817
// MI455X (gfx1250) — compile-verified, matrix operand order checked by reading, not run on hardware
//
#include <hip/hip_runtime.h>
#include <hip/hip_bf16.h>

typedef __attribute__((ext_vector_type(16))) _Float16 v16h;
typedef __attribute__((ext_vector_type(8)))  float    v8f;
typedef __attribute__((ext_vector_type(4)))  float    v4f;

#define SEQ     512
#define BATCH   128
#define HID     512
#define NCOL    2048
#define BH      (BATCH*HID)
#define NBLOCKS 64
#define GRP_BLKS 8                    // blocks per mtile sync group
#define PH_ELEMS (8*16*32*16)         // packed-h halves per ping-pong buffer
#define PW_ELEMS ((size_t)128*32*32*16)
#define XP_ELEMS ((size_t)SEQ*8*16*32*16)   // prepacked x halves (64 MB)

// CDNA5 16-bit fragment K mapping (A rows / B cols identical):
//   lane<16 : j=0..7 -> K=0..7,  j=8..15 -> K=16..23
//   lane>=16: j=0..7 -> K=8..15, j=8..15 -> K=24..31
__device__ __forceinline__ int klocal(int j, int half) {
    return (j & 7) + ((j >> 3) << 4) + (half << 3);
}

struct BFrag { v16h b0, b1, b2, b3; };

// WGP-scope prefetch (SCOPE=0 default): pulls into ALL cache levels incl. WGP$.
// __builtin_prefetch lowers to scope:SCOPE_SE which skips the WGP cache (ISA 10.5),
// so use inline asm with the default (WGP) scope instead.
template <int OFF>
__device__ __forceinline__ void prefetch_wgp(const void* p) {
    asm volatile("global_prefetch_b8 %0, off offset:%1" :: "v"(p), "i"(OFF));
}

__device__ __forceinline__ float sigm(float x) {
    return __builtin_amdgcn_rcpf(1.0f + __builtin_amdgcn_exp2f(x * -1.4426950408889634f));
}
__device__ __forceinline__ float tanh_(float x) {
    return 2.0f * __builtin_amdgcn_rcpf(1.0f + __builtin_amdgcn_exp2f(x * -2.8853900817779268f)) - 1.0f;
}

// ---- one-time weight pack: [nTile(128)][kc(32)][lane(32)][16] f16 ----
__global__ void __launch_bounds__(256)
lstm_pack_weights(const float* __restrict__ W, const float* __restrict__ b,
                  _Float16* __restrict__ packedW, float* __restrict__ biasT,
                  unsigned* __restrict__ counters) {
    int gid = blockIdx.x * blockDim.x + threadIdx.x;       // 131072
    if (gid >= 128 * 32 * 32) return;
    int lane  = gid & 31;
    int kc    = (gid >> 5) & 31;
    int nTile = gid >> 10;
    int col   = nTile * 16 + (lane & 15);
    int half  = lane >> 4;
    int rowBase = (kc < 16) ? (kc * 32) : (513 + (kc - 16) * 32);
    _Float16* dst = packedW + (size_t)gid * 16;
#pragma unroll
    for (int j = 0; j < 16; ++j)
        dst[j] = (_Float16)W[(size_t)(rowBase + klocal(j, half)) * NCOL + col];
    if (gid < NCOL) biasT[gid] = b[gid] + W[(size_t)512 * NCOL + gid];
    if (gid < 256)  counters[gid] = 0u;
}

// ---- one-time x pack: [t(512)][mtile(8)][kc(16)][lane(32)][16] f16 ----
__global__ void __launch_bounds__(256)
lstm_pack_x(const float* __restrict__ x, _Float16* __restrict__ xpacked) {
    size_t gid = (size_t)blockIdx.x * blockDim.x + threadIdx.x;  // 2,097,152
    if (gid >= XP_ELEMS / 16) return;
    int lane = (int)(gid & 31);
    int kc   = (int)((gid >> 5) & 15);
    int mt   = (int)((gid >> 9) & 7);
    int t    = (int)(gid >> 12);
    int l16  = lane & 15;
    int half = lane >> 4;
    const float* arow = x + (size_t)t * BH + (size_t)(mt * 16 + l16) * HID + kc * 32 + half * 8;
    v4f a0 = *(const v4f*)(arow +  0);
    v4f a1 = *(const v4f*)(arow +  4);
    v4f a2 = *(const v4f*)(arow + 16);
    v4f a3 = *(const v4f*)(arow + 20);
    v16h a;
    a[0]=(_Float16)a0.x; a[1]=(_Float16)a0.y; a[2]=(_Float16)a0.z; a[3]=(_Float16)a0.w;
    a[4]=(_Float16)a1.x; a[5]=(_Float16)a1.y; a[6]=(_Float16)a1.z; a[7]=(_Float16)a1.w;
    a[8]=(_Float16)a2.x; a[9]=(_Float16)a2.y; a[10]=(_Float16)a2.z; a[11]=(_Float16)a2.w;
    a[12]=(_Float16)a3.x; a[13]=(_Float16)a3.y; a[14]=(_Float16)a3.z; a[15]=(_Float16)a3.w;
    *(v16h*)(xpacked + gid * 16) = a;
}

// ---- persistent LSTM: 64 blocks x 128 thr = 256 waves; wave=(mtile, jt) ----
template <bool XP>
__global__ void __launch_bounds__(128)
lstm_persistent(const float* __restrict__ x,
                const _Float16* __restrict__ xpacked,
                const _Float16* __restrict__ packedW,
                const float* __restrict__ biasT,
                float* __restrict__ c_state,
                float* __restrict__ out,
                _Float16* __restrict__ packedH,
                unsigned* __restrict__ counters)
{
    const int lane  = threadIdx.x & 31;
    const int wid   = blockIdx.x * 4 + (threadIdx.x >> 5);
    const int mtile = wid >> 5;
    const int jt    = wid & 31;
    const int l16   = lane & 15;
    const int half  = lane >> 4;

    const size_t nStride = (size_t)32 * 32 * 16;
    const _Float16* Wb0 = packedW + (size_t)(0*32 + jt) * nStride + (size_t)lane * 16;
    const _Float16* Wb1 = packedW + (size_t)(1*32 + jt) * nStride + (size_t)lane * 16;
    const _Float16* Wb2 = packedW + (size_t)(2*32 + jt) * nStride + (size_t)lane * 16;
    const _Float16* Wb3 = packedW + (size_t)(3*32 + jt) * nStride + (size_t)lane * 16;

    const int col = jt * 16 + l16;
    const float bf = biasT[0 * HID + col];
    const float bi = biasT[1 * HID + col];
    const float bg = biasT[2 * HID + col];
    const float bo = biasT[3 * HID + col];

    // packed-h scatter slot (t-invariant)
    const int klcl  = ((jt & 1) << 4) + l16;
    const int half2 = (klcl >> 3) & 1;
    const int jj    = (klcl & 7) + ((klcl >> 4) << 3);
    const size_t ph_off = ((size_t)(mtile * 16 + (jt >> 1)) * 32 + half2 * 16 + half * 8) * 16 + jj;

    unsigned* myctr = counters + (blockIdx.x >> 3) * 32;   // per-mtile counter, own line

    auto loadB = [&](int kc) -> BFrag {
        BFrag f;
        const size_t o = (size_t)kc * 512;
        f.b0 = *(const v16h*)(Wb0 + o);
        f.b1 = *(const v16h*)(Wb1 + o);
        f.b2 = *(const v16h*)(Wb2 + o);
        f.b3 = *(const v16h*)(Wb3 + o);
        return f;
    };

    float cf[8];
#pragma unroll
    for (int r = 0; r < 8; ++r) cf[r] = 0.0f;

    // Prefetch t=0 x fragments into WGP$ (16 KB packed; 32 KB raw f32).
    if (XP) {
        const char* p0 = (const char*)(xpacked + ((size_t)mtile * 16) * 32 * 16)
                         + (size_t)lane * 128;
        prefetch_wgp<0>(p0);       prefetch_wgp<4096>(p0);
        prefetch_wgp<8192>(p0);    prefetch_wgp<12288>(p0);
    } else {
        const char* p0 = (const char*)(x + (size_t)mtile * 16 * HID) + (size_t)lane * 128;
        prefetch_wgp<0>(p0);       prefetch_wgp<4096>(p0);
        prefetch_wgp<8192>(p0);    prefetch_wgp<12288>(p0);
        prefetch_wgp<16384>(p0);   prefetch_wgp<20480>(p0);
        prefetch_wgp<24576>(p0);   prefetch_wgp<28672>(p0);
    }

    for (int t = 0; t < SEQ; ++t) {
        const float*    x_t   = x + (size_t)t * BH;
        const _Float16* xp_t  = xpacked + (((size_t)(t * 8 + mtile) * 16) * 32 + lane) * 16;
        const _Float16* ph_rd = packedH + (size_t)((t + 1) & 1) * PH_ELEMS
                                + ((size_t)mtile * 16 * 32 + lane) * 16;
        _Float16*       ph_wr = packedH + (size_t)(t & 1) * PH_ELEMS;
        float*          h_out = out + (size_t)t * BH;

        // Prefetch this step's h-fragment region into WGP$ (valid since last
        // barrier); the x-half runs first, giving ~16 chunks of lead time.
        if (t != 0) {
            const char* ph0 = (const char*)(packedH + (size_t)((t + 1) & 1) * PH_ELEMS
                                            + (size_t)mtile * 16 * 32 * 16)
                              + (size_t)lane * 128;
            prefetch_wgp<0>(ph0);      prefetch_wgp<4096>(ph0);
            prefetch_wgp<8192>(ph0);   prefetch_wgp<12288>(ph0);
        }
        // Prefetch next step's x fragments (read-only; a full step of lead time).
        if (t + 1 < SEQ) {
            if (XP) {
                const char* xn = (const char*)(xpacked
                                 + (((size_t)(t + 1) * 8 + mtile) * 16) * 32 * 16)
                                 + (size_t)lane * 128;
                prefetch_wgp<0>(xn);      prefetch_wgp<4096>(xn);
                prefetch_wgp<8192>(xn);   prefetch_wgp<12288>(xn);
            } else {
                const char* xn = (const char*)(x + (size_t)(t + 1) * BH
                                 + (size_t)mtile * 16 * HID) + (size_t)lane * 128;
                prefetch_wgp<0>(xn);      prefetch_wgp<4096>(xn);
                prefetch_wgp<8192>(xn);   prefetch_wgp<12288>(xn);
                prefetch_wgp<16384>(xn);  prefetch_wgp<20480>(xn);
                prefetch_wgp<24576>(xn);  prefetch_wgp<28672>(xn);
            }
        }

        auto loadAx = [&](int kc) -> v16h {
            if (XP) {
                return *(const v16h*)(xp_t + (size_t)kc * 512);
            } else {
                const float* arow = x_t + (size_t)(mtile * 16 + l16) * HID + kc * 32 + half * 8;
                v4f a0 = *(const v4f*)(arow +  0);
                v4f a1 = *(const v4f*)(arow +  4);
                v4f a2 = *(const v4f*)(arow + 16);
                v4f a3 = *(const v4f*)(arow + 20);
                v16h a;
                a[0]=(_Float16)a0.x; a[1]=(_Float16)a0.y; a[2]=(_Float16)a0.z; a[3]=(_Float16)a0.w;
                a[4]=(_Float16)a1.x; a[5]=(_Float16)a1.y; a[6]=(_Float16)a1.z; a[7]=(_Float16)a1.w;
                a[8]=(_Float16)a2.x; a[9]=(_Float16)a2.y; a[10]=(_Float16)a2.z; a[11]=(_Float16)a2.w;
                a[12]=(_Float16)a3.x; a[13]=(_Float16)a3.y; a[14]=(_Float16)a3.z; a[15]=(_Float16)a3.w;
                return a;
            }
        };
        auto loadAh = [&](int i) -> v16h {             // i = 0..15
            return *(const v16h*)(ph_rd + (size_t)i * 512);
        };

        v8f acc0 = {}, acc1 = {}, acc2 = {}, acc3 = {};
        auto wmma4 = [&](const v16h& a, const BFrag& bb) {
            acc0 = __builtin_amdgcn_wmma_f32_16x16x32_f16(false, a, false, bb.b0, (short)0, acc0, false, false);
            acc1 = __builtin_amdgcn_wmma_f32_16x16x32_f16(false, a, false, bb.b1, (short)0, acc1, false, false);
            acc2 = __builtin_amdgcn_wmma_f32_16x16x32_f16(false, a, false, bb.b2, (short)0, acc2, false, false);
            acc3 = __builtin_amdgcn_wmma_f32_16x16x32_f16(false, a, false, bb.b3, (short)0, acc3, false, false);
        };

        // ---- x half: straight-line, fully unrolled, 1-deep A pipeline ----
        v16h aC  = loadAx(0);
        BFrag bC = loadB(0);
#pragma unroll
        for (int kc = 0; kc < 15; ++kc) {
            v16h aN  = loadAx(kc + 1);
            BFrag bN = loadB(kc + 1);
            wmma4(aC, bC);
            aC = aN; bC = bN;
        }
        if (t != 0) {
            v16h aN  = loadAh(0);
            BFrag bN = loadB(16);
            wmma4(aC, bC);
            aC = aN; bC = bN;
            // ---- h half ----
#pragma unroll
            for (int i = 0; i < 15; ++i) {
                v16h aN2  = loadAh(i + 1);
                BFrag bN2 = loadB(17 + i);
                wmma4(aC, bC);
                aC = aN2; bC = bN2;
            }
            wmma4(aC, bC);
        } else {
            wmma4(aC, bC);
        }

        // ---- epilogue: D elem r -> (M = mtile*16 + half*8 + r, N = col) ----
#pragma unroll
        for (int r = 0; r < 8; ++r) {
            const int m = mtile * 16 + half * 8 + r;
            const size_t idx = (size_t)m * HID + col;
            const float f = sigm (acc0[r] + bf);
            const float i = sigm (acc1[r] + bi);
            const float g = tanh_(acc2[r] + bg);
            const float o = sigm (acc3[r] + bo);
            cf[r] = f * cf[r] + i * g;                 // cell state lives in VGPRs
            const float hn = o * tanh_(cf[r]);
            __builtin_nontemporal_store(hn, &h_out[idx]);
            ph_wr[ph_off + (size_t)r * 16] = (_Float16)hn;
        }

        // ---- per-mtile-group barrier (8 blocks, monotonic counter) ----
        __threadfence();                               // publish ph stores (all threads)
        __syncthreads();
        if (threadIdx.x == 0) {
            __hip_atomic_fetch_add(myctr, 1u, __ATOMIC_RELEASE, __HIP_MEMORY_SCOPE_AGENT);
            const unsigned tgt = (unsigned)(t + 1) * GRP_BLKS;
            while (__hip_atomic_load(myctr, __ATOMIC_ACQUIRE, __HIP_MEMORY_SCOPE_AGENT) < tgt)
                __builtin_amdgcn_s_sleep(1);
        }
        __syncthreads();
        __threadfence();                               // acquire (all threads)
    }

    // write final cell state once
#pragma unroll
    for (int r = 0; r < 8; ++r) {
        const int m = mtile * 16 + half * 8 + r;
        c_state[(size_t)m * HID + col] = cf[r];
    }
}

extern "C" void kernel_launch(void* const* d_in, const int* in_sizes, int n_in,
                              void* d_out, int out_size, void* d_ws, size_t ws_size,
                              hipStream_t stream) {
    const float* x = (const float*)d_in[0];   // (512, 128, 512)
    const float* W = (const float*)d_in[1];   // (1025, 2048)
    const float* b = (const float*)d_in[2];   // (2048,)
    float* out = (float*)d_out;

    char* p = (char*)d_ws;
    _Float16* packedW = (_Float16*)p;            p += PW_ELEMS * sizeof(_Float16);  // 4 MB
    float*    biasT   = (float*)p;               p += NCOL * sizeof(float);
    float*    c_state = (float*)p;               p += (size_t)BH * sizeof(float);
    _Float16* packedH = (_Float16*)p;            p += (size_t)2 * PH_ELEMS * sizeof(_Float16);
    unsigned* counters= (unsigned*)p;            p += 256 * sizeof(unsigned);
    _Float16* xpacked = (_Float16*)p;
    const size_t need_xp = (size_t)(p - (char*)d_ws) + XP_ELEMS * sizeof(_Float16);
    const bool use_xp = (ws_size >= need_xp);

    lstm_pack_weights<<<512, 256, 0, stream>>>(W, b, packedW, biasT, counters);
    if (use_xp) {
        lstm_pack_x<<<(int)((XP_ELEMS / 16 + 255) / 256), 256, 0, stream>>>(x, xpacked);
        lstm_persistent<true ><<<NBLOCKS, 128, 0, stream>>>(x, xpacked, packedW, biasT,
                                                            c_state, out, packedH, counters);
    } else {
        lstm_persistent<false><<<NBLOCKS, 128, 0, stream>>>(x, xpacked, packedW, biasT,
                                                            c_state, out, packedH, counters);
    }

    // tail: h_last = h[511], c_last = final cell state
    float* tail = out + (size_t)SEQ * BH;
    hipMemcpyAsync(tail,      out + (size_t)(SEQ - 1) * BH, BH * sizeof(float),
                   hipMemcpyDeviceToDevice, stream);
    hipMemcpyAsync(tail + BH, c_state,                      BH * sizeof(float),
                   hipMemcpyDeviceToDevice, stream);
}
